// GTNet_decoder_88527865905291
// MI455X (gfx1250) — compile-verified, matrix operand order checked by reading, not run on hardware
//
#include <hip/hip_runtime.h>
#include <hip/hip_bf16.h>

// ---------------------------------------------------------------------------
// Types for WMMA
// ---------------------------------------------------------------------------
typedef __attribute__((ext_vector_type(16))) _Float16 v16h;
typedef __attribute__((ext_vector_type(8)))  _Float16 v8h;
typedef __attribute__((ext_vector_type(4)))  _Float16 v4h;
typedef __attribute__((ext_vector_type(8)))  float    v8f;

#define TILE_M 32
#define TILE_N 128
#define TILE_K 64
#define LDS_LD 72   // halves per LDS row: 144 B, multiple of 16 B, bank-spreading

// Activation codes
#define ACT_NONE   0
#define ACT_LRELU  1
#define ACT_RELU   2
#define ACT_SIGM   3

// ---------------------------------------------------------------------------
// WMMA GEMM:  Y(O,S) = act( W(O,C) * X(C,S) + bias + R )
// fp32 in/out, f16 WMMA compute with f32 accumulation.
// Block = 256 threads = 8 waves (2x4); each wave owns a 16x32 output strip
// (two 16x16 fragments sharing A) -> 4 x v_wmma_f32_16x16x32_f16 per wave
// per K=64 staging round.
// Staging: block-uniform fast path (fully in-bounds tile -> unconditional
// batched loads); slow path uses clamp + mask-multiply so loads cannot be
// sunk into divergent blocks.
// ---------------------------------------------------------------------------
__global__ __launch_bounds__(256) void k_gemm(
    const float* __restrict__ W, const float* __restrict__ X,
    const float* __restrict__ bias, const float* __restrict__ R,
    float* __restrict__ Y, int O, int C, int S,
    long long xbs, long long ybs, long long rbs, int act, float alpha)
{
  __shared__ alignas(16) _Float16 Ws[TILE_M][LDS_LD];
  __shared__ alignas(16) _Float16 Xs[TILE_N][LDS_LD];   // transposed: Xs[n][k]

  const int b = blockIdx.z;
  const float* Xb = X + (long long)b * xbs;
  float*       Yb = Y + (long long)b * ybs;
  const float* Rb = R ? (R + (long long)b * rbs) : nullptr;

  const int tm = blockIdx.y * TILE_M;
  const int tn = blockIdx.x * TILE_N;

  const int tid  = threadIdx.x;
  const int lane = tid & 31;
  const int wave = tid >> 5;
  const int wm   = wave >> 2;         // 0..1 (wave row, 16 rows each)
  const int wn   = wave & 3;          // 0..3 (wave col, 32 cols each)
  const int idx16 = lane & 15;        // M index for A, N index for B
  const int g8    = (lane >> 4) << 3; // 0 or 8 (K sub-group per ISA layout)

  // W staging: 32 rows x 16 quad-groups = 512 quads -> 2 per thread
  const int wr0 = tid >> 4;           // row item0 (0..15), item1 adds 16
  const int wkq = (tid & 15) << 2;    // k quad offset 0..60
  // X staging: 128 cols x 16 quad-groups = 2048 quads -> 8 per thread
  const int xn   = tid & 127;         // n
  const int xkg0 = tid >> 7;          // 0 or 1 ; item j adds 2*j

  v8f acc[2] = { {}, {} };

  for (int k0 = 0; k0 < C; k0 += TILE_K) {
    const bool fullK = (k0 + TILE_K <= C);
    // ---- stage W tile ----
    if (fullK && (tm + TILE_M <= O)) {
      #pragma unroll
      for (int j = 0; j < 2; ++j) {
        int r = wr0 + j * 16;
        const float* wp = W + (long long)(tm + r) * C + k0 + wkq;
        v4h h = { (_Float16)wp[0], (_Float16)wp[1], (_Float16)wp[2], (_Float16)wp[3] };
        *(v4h*)&Ws[r][wkq] = h;
      }
    } else {
      #pragma unroll
      for (int j = 0; j < 2; ++j) {
        int r  = wr0 + j * 16;
        int go = tm + r;
        int goc = (go < O) ? go : (O - 1);
        const float* wp = W + (long long)goc * C;
        float f[4];
        #pragma unroll
        for (int u = 0; u < 4; ++u) {
          int gk  = k0 + wkq + u;
          int gkc = (gk < C) ? gk : (C - 1);
          float msk = (go < O && gk < C) ? 1.0f : 0.0f;
          f[u] = wp[gkc] * msk;           // mask-multiply: load cannot be sunk
        }
        v4h h = { (_Float16)f[0], (_Float16)f[1], (_Float16)f[2], (_Float16)f[3] };
        *(v4h*)&Ws[r][wkq] = h;
      }
    }
    // ---- stage X tile transposed (Xs[n][k]) ----
    if (fullK && (tn + TILE_N <= S)) {
      #pragma unroll
      for (int j = 0; j < 8; ++j) {
        int kq = (xkg0 + 2 * j) << 2;
        const float* xp = Xb + (long long)(k0 + kq) * S + tn + xn;
        v4h h = { (_Float16)xp[0], (_Float16)xp[(long long)S],
                  (_Float16)xp[2LL * S], (_Float16)xp[3LL * S] };
        *(v4h*)&Xs[xn][kq] = h;
      }
    } else {
      #pragma unroll
      for (int j = 0; j < 8; ++j) {
        int kq  = (xkg0 + 2 * j) << 2;
        int gs  = tn + xn;
        int gsc = (gs < S) ? gs : (S - 1);
        float f[4];
        #pragma unroll
        for (int u = 0; u < 4; ++u) {
          int gk  = k0 + kq + u;
          int gkc = (gk < C) ? gk : (C - 1);
          float msk = (gk < C && gs < S) ? 1.0f : 0.0f;
          f[u] = Xb[(long long)gkc * S + gsc] * msk;
        }
        v4h h = { (_Float16)f[0], (_Float16)f[1], (_Float16)f[2], (_Float16)f[3] };
        *(v4h*)&Xs[xn][kq] = h;
      }
    }
    if (k0 + TILE_K < C)   // prefetch next X tile (global_prefetch_b8)
      __builtin_prefetch(Xb + (long long)(k0 + TILE_K) * S + tn + xn, 0, 1);
    __syncthreads();

    // ---- A fragments (shared by both column fragments) ----
    const _Float16* ap = &Ws[wm * 16 + idx16][0];
    v8h a0 = *(const v8h*)(ap + g8);
    v8h a1 = *(const v8h*)(ap + g8 + 16);
    v8h a2 = *(const v8h*)(ap + g8 + 32);
    v8h a3 = *(const v8h*)(ap + g8 + 48);
    v16h A0 = __builtin_shufflevector(a0, a1, 0,1,2,3,4,5,6,7,8,9,10,11,12,13,14,15);
    v16h A1 = __builtin_shufflevector(a2, a3, 0,1,2,3,4,5,6,7,8,9,10,11,12,13,14,15);
    // ---- two B fragments, 4 WMMA ----
    #pragma unroll
    for (int cfr = 0; cfr < 2; ++cfr) {
      const _Float16* bp = &Xs[wn * 32 + cfr * 16 + idx16][0];
      v8h b0 = *(const v8h*)(bp + g8);
      v8h b1 = *(const v8h*)(bp + g8 + 16);
      v8h b2 = *(const v8h*)(bp + g8 + 32);
      v8h b3 = *(const v8h*)(bp + g8 + 48);
      v16h B0 = __builtin_shufflevector(b0, b1, 0,1,2,3,4,5,6,7,8,9,10,11,12,13,14,15);
      v16h B1 = __builtin_shufflevector(b2, b3, 0,1,2,3,4,5,6,7,8,9,10,11,12,13,14,15);
      acc[cfr] = __builtin_amdgcn_wmma_f32_16x16x32_f16(false, A0, false, B0,
                                                        (short)0, acc[cfr], false, false);
      acc[cfr] = __builtin_amdgcn_wmma_f32_16x16x32_f16(false, A1, false, B1,
                                                        (short)0, acc[cfr], false, false);
    }
    __syncthreads();
  }

  // ---- epilogue: C/D layout -> vgpr r holds M = r + 8*(lane>=16), N = lane&15 ----
  #pragma unroll
  for (int cfr = 0; cfr < 2; ++cfr) {
    int col = tn + wn * 32 + cfr * 16 + idx16;
    #pragma unroll
    for (int r = 0; r < 8; ++r) {
      int row = tm + wm * 16 + r + ((lane >> 4) << 3);
      if (row < O && col < S) {
        float v = acc[cfr][r];
        if (bias) v += bias[row];
        if (Rb)   v += Rb[(long long)row * S + col];
        if      (act == ACT_LRELU) v = (v >= 0.0f) ? v : alpha * v;
        else if (act == ACT_RELU)  v = fmaxf(v, 0.0f);
        else if (act == ACT_SIGM)  v = 1.0f / (1.0f + __expf(-v));
        Yb[(long long)row * S + col] = v;
      }
    }
  }
}

// ---------------------------------------------------------------------------
// GPool top-k (k=1024 of N=2048) by exact rank counting; sorted descending,
// lower index wins ties (matches lax.top_k).  One block per batch.
// ---------------------------------------------------------------------------
__global__ __launch_bounds__(256) void k_topk(const float* __restrict__ score,
                                              int* __restrict__ idx,
                                              float* __restrict__ sk, int N, int K)
{
  int b = blockIdx.x;
  const float* s = score + (long long)b * N;
  __shared__ float ls[2048];
  for (int i = threadIdx.x; i < N; i += 256) ls[i] = s[i];
  __syncthreads();
  for (int i = threadIdx.x; i < N; i += 256) {
    float si = ls[i];
    int r = 0;
    for (int j = 0; j < N; ++j) {
      float sj = ls[j];
      r += (sj > si) || (sj == si && j < i);
    }
    if (r < K) { idx[(long long)b * K + r] = i; sk[(long long)b * K + r] = si; }
  }
}

// Gather columns by index, optional score scaling: dst[c][j] = src[c][idx[j]] * (sk[j]?)
__global__ void k_gather(float* __restrict__ dst, const float* __restrict__ src,
                         const int* __restrict__ idx, const float* __restrict__ sk,
                         int C, int Sin, int K, int scale, long long dbs, long long sbs)
{
  int b = blockIdx.z;
  long long t = (long long)blockIdx.x * 256 + threadIdx.x;
  if (t >= (long long)C * K) return;
  int c = (int)(t / K), j = (int)(t % K);
  int id = idx[(long long)b * K + j];
  float v = src[b * sbs + (long long)c * Sin + id];
  if (scale) v *= sk[(long long)b * K + j];
  dst[b * dbs + (long long)c * K + j] = v;
}

// Row-max over spatial: vec[b][c] = max_s X[b][c][s]
__global__ __launch_bounds__(256) void k_rowmax(const float* __restrict__ x,
                                                float* __restrict__ v,
                                                int S, long long xbs, int C)
{
  int c = blockIdx.x, b = blockIdx.y;
  const float* p = x + b * xbs + (long long)c * S;
  float m = -3.4e38f;
  for (int i = threadIdx.x; i < S; i += 256) m = fmaxf(m, p[i]);
  __shared__ float red[256];
  red[threadIdx.x] = m; __syncthreads();
  for (int off = 128; off > 0; off >>= 1) {
    if (threadIdx.x < off) red[threadIdx.x] = fmaxf(red[threadIdx.x], red[threadIdx.x + off]);
    __syncthreads();
  }
  if (threadIdx.x == 0) v[(long long)b * C + c] = red[0];
}

// dst[b][c][s] = src[b][c][s]  (2D block copy with independent strides)
__global__ void k_copy2d(float* __restrict__ dst, long long dbs, int dld,
                         const float* __restrict__ src, long long sbs, int sld,
                         int C, int S)
{
  int b = blockIdx.z;
  long long t = (long long)blockIdx.x * 256 + threadIdx.x;
  if (t >= (long long)C * S) return;
  int c = (int)(t / S), s = (int)(t % S);
  dst[b * dbs + (long long)c * dld + s] = src[b * sbs + (long long)c * sld + s];
}

// dst[b][c][s] = vec[b][c]
__global__ void k_bcastrows(float* __restrict__ dst, long long dbs, int dld,
                            const float* __restrict__ vec, int vbs, int C, int S)
{
  int b = blockIdx.z;
  long long t = (long long)blockIdx.x * 256 + threadIdx.x;
  if (t >= (long long)C * S) return;
  int c = (int)(t / S), s = (int)(t % S);
  dst[b * dbs + (long long)c * dld + s] = vec[(long long)b * vbs + c];
}

// Tiny FC: y[b][o] = act( sum_c W[o][c]*x[b][c] + bias[o] )
__global__ void k_fc(const float* __restrict__ W, const float* __restrict__ bias,
                     const float* __restrict__ x, float* __restrict__ y,
                     int O, int C, int act, float alpha)
{
  int b = blockIdx.x, o = threadIdx.x;
  if (o >= O) return;
  const float* xb = x + (long long)b * C;
  float acc = bias ? bias[o] : 0.0f;
  for (int c = 0; c < C; ++c) acc += W[(long long)o * C + c] * xb[c];
  if (act == ACT_LRELU) acc = (acc >= 0.0f) ? acc : alpha * acc;
  y[(long long)b * O + o] = acc;
}

// mis[b][d][m] = sum_c pk[b][c][m] * T[b][c*3+d]
__global__ void k_apply_T(const float* __restrict__ pk, const float* __restrict__ T,
                          float* __restrict__ mis, int M)
{
  int b = blockIdx.z;
  int m = blockIdx.x * 256 + threadIdx.x;
  if (m >= M) return;
  const float* p = pk + (long long)b * 3 * M;
  const float* t = T + (long long)b * 9;
  float x = p[m], y = p[M + m], z = p[2 * M + m];
  float* o = mis + (long long)b * 3 * M;
  o[m]         = x * t[0] + y * t[3] + z * t[6];
  o[M + m]     = x * t[1] + y * t[4] + z * t[7];
  o[2 * M + m] = x * t[2] + y * t[5] + z * t[8];
}

// ---------------------------------------------------------------------------
// KNN top-32 of N points per query; register-resident sorted list (ascending).
// One thread per query, point cloud tiled through LDS.
// ---------------------------------------------------------------------------
#define KNN_K 32
__global__ __launch_bounds__(128) void k_knn(const float* __restrict__ xyz,  // (3,N)
                                             const float* __restrict__ q,    // (3,M)
                                             int* __restrict__ kidx, int N, int M)
{
  int m = blockIdx.x * blockDim.x + threadIdx.x;
  float qx = 0.f, qy = 0.f, qz = 0.f;
  if (m < M) { qx = q[m]; qy = q[M + m]; qz = q[2 * M + m]; }
  float bd[KNN_K]; int bi[KNN_K];
  #pragma unroll
  for (int t = 0; t < KNN_K; ++t) { bd[t] = 3.4e38f; bi[t] = 0; }

  __shared__ float px[256], py[256], pz[256];
  for (int base = 0; base < N; base += 256) {
    for (int i = threadIdx.x; i < 256; i += blockDim.x) {
      px[i] = xyz[base + i]; py[i] = xyz[N + base + i]; pz[i] = xyz[2 * N + base + i];
    }
    __syncthreads();
    if (m < M) {
      for (int i = 0; i < 256; ++i) {
        float dx = px[i] - qx, dy = py[i] - qy, dz = pz[i] - qz;
        float d = dx * dx + dy * dy + dz * dz;
        if (d < bd[KNN_K - 1]) {
          unsigned le = 0u;                     // bit t: bd[t] <= d (sorted -> low-bit run)
          #pragma unroll
          for (int t = 0; t < KNN_K; ++t) le |= (bd[t] <= d) ? (1u << t) : 0u;
          #pragma unroll
          for (int t = KNN_K - 1; t >= 1; --t)  // shift elements > d down by one
            if (!((le >> (t - 1)) & 1u)) { bd[t] = bd[t - 1]; bi[t] = bi[t - 1]; }
          #pragma unroll
          for (int t = 0; t < KNN_K; ++t) {     // insert at first slot with bd[t] > d
            bool here = !((le >> t) & 1u) && (t == 0 || ((le >> (t - 1)) & 1u));
            if (here) { bd[t] = d; bi[t] = base + i; }
          }
        }
      }
    }
    __syncthreads();
  }
  if (m < M) {
    #pragma unroll
    for (int t = 0; t < KNN_K; ++t) kidx[(long long)m * KNN_K + t] = bi[t];
  }
}

// Build 10-channel KNN features: rows [diff(3), ctr(3), nbr(3), dist(1)], s = kk*M+m
__global__ void k_knn_feat(const float* __restrict__ xyz, const float* __restrict__ q,
                           const int* __restrict__ kidx, float* __restrict__ f,
                           int N, int M)
{
  long long t = (long long)blockIdx.x * 256 + threadIdx.x;
  long long KM = (long long)M * KNN_K;
  if (t >= KM) return;
  int kk = (int)(t / M), m = (int)(t % M);
  int id = kidx[(long long)m * KNN_K + kk];
  float cx = q[m], cy = q[M + m], cz = q[2 * M + m];
  float nx = xyz[id], ny = xyz[N + id], nz = xyz[2 * N + id];
  float dx = nx - cx, dy = ny - cy, dz = nz - cz;
  float dist = sqrtf(dx * dx + dy * dy + dz * dz + 1e-12f);
  f[0 * KM + t] = dx; f[1 * KM + t] = dy; f[2 * KM + t] = dz;
  f[3 * KM + t] = cx; f[4 * KM + t] = cy; f[5 * KM + t] = cz;
  f[6 * KM + t] = nx; f[7 * KM + t] = ny; f[8 * KM + t] = nz;
  f[9 * KM + t] = dist;
}

// out[c][m] = max_kk in[c][kk*M+m]
__global__ void k_max_over_k(const float* __restrict__ in, float* __restrict__ out,
                             int C, int M, int K)
{
  long long t = (long long)blockIdx.x * 256 + threadIdx.x;
  if (t >= (long long)C * M) return;
  int c = (int)(t / M), m = (int)(t % M);
  const float* p = in + (long long)c * M * K;
  float v = -3.4e38f;
  for (int kk = 0; kk < K; ++kk) v = fmaxf(v, p[(long long)kk * M + m]);
  out[(long long)c * M + m] = v;
}

// dst[c][kk*M+m] = src[c][m]   (dst pre-offset to its row region)
__global__ void k_bcast_over_k(float* __restrict__ dst, const float* __restrict__ src,
                               int C, int M, int K)
{
  long long t = (long long)blockIdx.x * 256 + threadIdx.x;
  long long tot = (long long)C * M * K;
  if (t >= tot) return;
  int c = (int)(t / ((long long)M * K));
  long long r = t % ((long long)M * K);
  int m = (int)(r % M);
  dst[t] = src[(long long)c * M + m];
}

// ---------------------------------------------------------------------------
// Host-side parameter walking & launch helpers
// ---------------------------------------------------------------------------
struct Lin   { const float *w, *b; };
struct MlpRes{ Lin sc, c1, c2a, c2b; };
struct PosDisp { MlpRes c1, c2; Lin c3; };
struct Gpool { const float *w1, *b1, *w2; };
struct Stn   { MlpRes c1, c2; Lin fc2, fc3; };
struct PmfP  { MlpRes cr1, cr2; };
struct Cur   { void* const* a; int i; const float* f() { return (const float*)a[i++]; } };

static Lin    getLin(Cur& c) { Lin l; l.w = c.f(); l.b = c.f(); return l; }
static MlpRes getMlp(Cur& c) { MlpRes m; m.sc = getLin(c); m.c1 = getLin(c); m.c2a = getLin(c); m.c2b = getLin(c); return m; }
static PosDisp getPd(Cur& c) { PosDisp p; p.c1 = getMlp(c); p.c2 = getMlp(c); p.c3 = getLin(c); return p; }

static inline dim3 g1(long long n, int bz = 1) { return dim3((unsigned)((n + 255) / 256), 1, (unsigned)bz); }

static void gemm(hipStream_t st, const float* W, const float* X, const float* bias,
                 const float* R, float* Y, int O, int C, int S, int Bz,
                 long long xbs, long long ybs, long long rbs, int act)
{
  dim3 g((unsigned)((S + TILE_N - 1) / TILE_N), (unsigned)((O + TILE_M - 1) / TILE_M), (unsigned)Bz);
  k_gemm<<<g, dim3(256), 0, st>>>(W, X, bias, R, Y, O, C, S, xbs, ybs, rbs, act, 0.2f);
}

// mlp_res: out = act( c2b(lrelu(c2a(lrelu(c1(x))))) + sc(x) )
static void mlpRes(hipStream_t st, const MlpRes& P, const float* X, long long xbs,
                   int Ci, int H, int Co, int S, int Bz,
                   float* t1, long long t1bs, float* t2, long long t2bs,
                   float* sc, long long scbs, float* out, long long obs, int actFinal)
{
  gemm(st, P.c1.w,  X,  P.c1.b,  nullptr, t1,  H,  Ci, S, Bz, xbs,  t1bs, 0,    ACT_LRELU);
  gemm(st, P.c2a.w, t1, P.c2a.b, nullptr, t2,  Co, H,  S, Bz, t1bs, t2bs, 0,    ACT_LRELU);
  gemm(st, P.sc.w,  X,  P.sc.b,  nullptr, sc,  Co, Ci, S, Bz, xbs,  scbs, 0,    ACT_NONE);
  gemm(st, P.c2b.w, t2, P.c2b.b, sc,      out, Co, Co, S, Bz, t2bs, obs,  scbs, actFinal);
}

// ---------------------------------------------------------------------------
// Full forward orchestration
// ---------------------------------------------------------------------------
extern "C" void kernel_launch(void* const* d_in, const int* in_sizes, int n_in,
                              void* d_out, int out_size, void* d_ws, size_t ws_size,
                              hipStream_t stream)
{
  (void)in_sizes; (void)n_in; (void)out_size; (void)ws_size;
  const int Bz = 4, N = 2048, M = 1024, Kn = KNN_K, KM = M * KNN_K;
  const int DG = 512, DL = 128, NPRIM = 8;

  const float* pos      = (const float*)d_in[0];   // (B,3,N)
  const float* f_local  = (const float*)d_in[1];   // (B,128,N)
  const float* f_global = (const float*)d_in[2];   // (B,512)
  Cur cur{d_in, 3};

  // ---- workspace bump allocator ----
  char* wsb = (char*)d_ws; size_t off = 0;
  auto aF = [&](size_t n) { float* p = (float*)(wsb + off); off += ((n * 4 + 255) / 256) * 256; return p; };
  auto aI = [&](size_t n) { int*   p = (int*)(wsb + off);   off += ((n * 4 + 255) / 256) * 256; return p; };

  float* glf   = aF((size_t)Bz * 640 * N);
  float* h160  = aF((size_t)Bz * 160 * N);
  float* score = aF((size_t)Bz * N);
  float* sk    = aF((size_t)Bz * M);
  int*   sidx  = aI((size_t)Bz * M);
  float* posk  = aF((size_t)Bz * 3 * M);
  float* posk2 = aF((size_t)Bz * 3 * M);
  float* flk   = aF((size_t)Bz * 128 * M);
  float* glfk  = aF((size_t)Bz * 640 * M);
  float* cc    = aF((size_t)Bz * 771 * M);
  float* t1    = aF((size_t)Bz * 128 * M);
  float* t2    = aF((size_t)Bz * 128 * M);
  float* t3    = aF((size_t)Bz * 128 * M);
  float* f128  = aF((size_t)Bz * 128 * M);
  float* vecA  = aF((size_t)Bz * 128);
  float* hv    = aF((size_t)Bz * 128);
  float* h2v   = aF((size_t)Bz * 64);
  float* Tm    = aF((size_t)Bz * 9);
  float* mis   = aF((size_t)Bz * 3 * M);
  float* mis2  = aF((size_t)Bz * 3 * M);
  float* mf    = aF((size_t)Bz * 128 * M);
  float* xm    = aF((size_t)64 * M);
  float* feat10= aF((size_t)10 * KM);
  int*   kidx  = aI((size_t)M * Kn);
  float* q1 = aF((size_t)128 * KM);
  float* q2 = aF((size_t)128 * KM);
  float* q3 = aF((size_t)128 * KM);
  float* q4 = aF((size_t)128 * KM);
  float* q5 = aF((size_t)128 * KM);

  const long long bs128 = 128LL * M, bs3 = 3LL * M, bs771 = 771LL * M, bs640 = 640LL * M;

  // ---- glf = concat(broadcast(f_global), f_local)  (B,640,N) ----
  k_bcastrows<<<g1((long long)DG * N, Bz), 256, 0, stream>>>(glf, 640LL * N, N, f_global, DG, DG, N);
  k_copy2d  <<<g1((long long)DL * N, Bz), 256, 0, stream>>>(glf + (long long)DG * N, 640LL * N, N,
                                                            f_local, 128LL * N, N, DL, N);

  float* cs = (float*)d_out;                            // (B,3,16*M)
  float* cf = cs + (long long)Bz * 3 * (2 * M * NPRIM); // (B,128,16*M)
  const int Wtot = 2 * M * NPRIM;

  for (int pidx = 0; pidx < NPRIM; ++pidx) {
    Gpool gp; gp.w1 = cur.f(); gp.b1 = cur.f(); gp.w2 = cur.f();
    PosDisp den = getPd(cur);
    Stn stn; stn.c1 = getMlp(cur); stn.c2 = getMlp(cur); stn.fc2 = getLin(cur); stn.fc3 = getLin(cur);
    PmfP pmf; pmf.cr1 = getMlp(cur); pmf.cr2 = getMlp(cur);
    PosDisp offp = getPd(cur);

    // ---- GPool: score + top-k + gathers ----
    gemm(stream, gp.w1, glf, gp.b1, nullptr, h160, 160, 640, N, Bz, 640LL * N, 160LL * N, 0, ACT_RELU);
    gemm(stream, gp.w2, h160, nullptr, nullptr, score, 1, 160, N, Bz, 160LL * N, (long long)N, 0, ACT_SIGM);
    k_topk<<<dim3(Bz), dim3(256), 0, stream>>>(score, sidx, sk, N, M);
    k_gather<<<g1(3LL * M, Bz), 256, 0, stream>>>(posk, pos, sidx, sk, 3, N, M, 0, bs3, 3LL * N);
    k_gather<<<g1(128LL * M, Bz), 256, 0, stream>>>(flk, f_local, sidx, sk, 128, N, M, 0, bs128, 128LL * N);
    k_gather<<<g1(640LL * M, Bz), 256, 0, stream>>>(glfk, glf, sidx, sk, 640, N, M, 1, bs640, 640LL * N);

    // ---- denoise = pos_displace(den, posk, glfk); posk2 = posk + disp ----
    mlpRes(stream, den.c1, posk, bs3, 3, 64, 128, M, Bz, t1, bs128, t2, bs128, t3, bs128, f128, bs128, ACT_NONE);
    k_rowmax<<<dim3(128, Bz), dim3(256), 0, stream>>>(f128, vecA, M, bs128, 128);
    k_copy2d  <<<g1(3LL * M, Bz), 256, 0, stream>>>(cc, bs771, M, posk, bs3, M, 3, M);
    k_bcastrows<<<g1(128LL * M, Bz), 256, 0, stream>>>(cc + 3LL * M, bs771, M, vecA, 128, 128, M);
    k_copy2d  <<<g1(640LL * M, Bz), 256, 0, stream>>>(cc + 131LL * M, bs771, M, glfk, bs640, M, 640, M);
    mlpRes(stream, den.c2, cc, bs771, 771, 128, 64, M, Bz, t1, bs128, t2, bs128, t3, bs128, f128, bs128, ACT_LRELU);
    gemm(stream, den.c3.w, f128, den.c3.b, posk, posk2, 3, 64, M, Bz, bs128, bs3, bs3, ACT_NONE);

    // ---- stn3d -> T (B,3,3), mis = pos_k^T T ----
    mlpRes(stream, stn.c1, posk2, bs3, 3, 64, 128, M, Bz, t1, bs128, t2, bs128, t3, bs128, f128, bs128, ACT_NONE);
    k_rowmax<<<dim3(128, Bz), dim3(256), 0, stream>>>(f128, vecA, M, bs128, 128);
    k_copy2d  <<<g1(3LL * M, Bz), 256, 0, stream>>>(cc, bs771, M, posk2, bs3, M, 3, M);
    k_bcastrows<<<g1(128LL * M, Bz), 256, 0, stream>>>(cc + 3LL * M, bs771, M, vecA, 128, 128, M);
    k_copy2d  <<<g1(640LL * M, Bz), 256, 0, stream>>>(cc + 131LL * M, bs771, M, glfk, bs640, M, 640, M);
    mlpRes(stream, stn.c2, cc, bs771, 771, 128, 128, M, Bz, t1, bs128, t2, bs128, t3, bs128, f128, bs128, ACT_NONE);
    k_rowmax<<<dim3(128, Bz), dim3(256), 0, stream>>>(f128, hv, M, bs128, 128);
    k_fc<<<dim3(Bz), dim3(64), 0, stream>>>(stn.fc2.w, stn.fc2.b, hv, h2v, 64, 128, ACT_LRELU, 0.2f);
    k_fc<<<dim3(Bz), dim3(64), 0, stream>>>(stn.fc3.w, stn.fc3.b, h2v, Tm, 9, 64, ACT_NONE, 0.2f);
    k_apply_T<<<g1(M, Bz), 256, 0, stream>>>(posk2, Tm, mis, M);

    // ---- point_mis_feat (per-batch to bound workspace) -> mf (B,128,M) ----
    for (int b = 0; b < Bz; ++b) {
      const float* xyz = pos + (long long)b * 3 * N;
      const float* qb  = mis + (long long)b * 3 * M;
      k_knn<<<dim3(M / 128), dim3(128), 0, stream>>>(xyz, qb, kidx, N, M);
      k_knn_feat<<<g1((long long)KM), 256, 0, stream>>>(xyz, qb, kidx, feat10, N, M);
      // cr1: 10 -> 64 -> 64 over S = KM
      mlpRes(stream, pmf.cr1, feat10, 0, 10, 64, 64, KM, 1, q1, 0, q2, 0, q3, 0, q4, 0, ACT_NONE);
      k_max_over_k<<<g1(64LL * M), 256, 0, stream>>>(q4, xm, 64, M, Kn);
      // concat [x ; broadcast(max_k x)] -> q1 (128, KM)
      k_copy2d<<<g1(64LL * KM), 256, 0, stream>>>(q1, 0, KM, q4, 0, KM, 64, KM);
      k_bcast_over_k<<<g1(64LL * KM), 256, 0, stream>>>(q1 + 64LL * KM, xm, 64, M, Kn);
      // cr2: 128 -> 64 -> 128
      mlpRes(stream, pmf.cr2, q1, 0, 128, 64, 128, KM, 1, q2, 0, q3, 0, q5, 0, q4, 0, ACT_NONE);
      k_max_over_k<<<g1(128LL * M), 256, 0, stream>>>(q4, mf + (long long)b * 128 * M, 128, M, Kn);
    }

    // ---- off = pos_displace(offp, mis, concat(f_global, mf)); mis2 = mis + disp ----
    mlpRes(stream, offp.c1, mis, bs3, 3, 64, 128, M, Bz, t1, bs128, t2, bs128, t3, bs128, f128, bs128, ACT_NONE);
    k_rowmax<<<dim3(128, Bz), dim3(256), 0, stream>>>(f128, vecA, M, bs128, 128);
    k_copy2d  <<<g1(3LL * M, Bz), 256, 0, stream>>>(cc, bs771, M, mis, bs3, M, 3, M);
    k_bcastrows<<<g1(128LL * M, Bz), 256, 0, stream>>>(cc + 3LL * M, bs771, M, vecA, 128, 128, M);
    k_bcastrows<<<g1(512LL * M, Bz), 256, 0, stream>>>(cc + 131LL * M, bs771, M, f_global, DG, 512, M);
    k_copy2d  <<<g1(128LL * M, Bz), 256, 0, stream>>>(cc + 643LL * M, bs771, M, mf, bs128, M, 128, M);
    mlpRes(stream, offp.c2, cc, bs771, 771, 128, 64, M, Bz, t1, bs128, t2, bs128, t3, bs128, f128, bs128, ACT_LRELU);
    gemm(stream, offp.c3.w, f128, offp.c3.b, mis, mis2, 3, 64, M, Bz, bs128, bs3, bs3, ACT_NONE);

    // ---- scatter results into output concat slices ----
    float* csp = cs + (long long)pidx * 2 * M;
    float* cfp = cf + (long long)pidx * 2 * M;
    k_copy2d<<<g1(3LL * M, Bz), 256, 0, stream>>>(csp,     3LL * Wtot,   Wtot, posk2, bs3,   M, 3,   M);
    k_copy2d<<<g1(3LL * M, Bz), 256, 0, stream>>>(csp + M, 3LL * Wtot,   Wtot, mis2,  bs3,   M, 3,   M);
    k_copy2d<<<g1(128LL * M, Bz), 256, 0, stream>>>(cfp,     128LL * Wtot, Wtot, flk, bs128, M, 128, M);
    k_copy2d<<<g1(128LL * M, Bz), 256, 0, stream>>>(cfp + M, 128LL * Wtot, Wtot, mf,  bs128, M, 128, M);
  }
}
